// ChebNet_62663572848803
// MI455X (gfx1250) — matrix operand order repaired and checked by reading, not compiled, not run on hardware
//
#include <hip/hip_runtime.h>

typedef __attribute__((ext_vector_type(16))) _Float16 v16h;
typedef __attribute__((ext_vector_type(8)))  float    v8f;

#define SCAN_B 256

// ---------------------------------------------------------------- utilities
__global__ void fill_zero32(unsigned int* __restrict__ p, int n) {
    int i = blockIdx.x * blockDim.x + threadIdx.x;
    if (i < n) p[i] = 0u;
}

__global__ void copy_i32(int* __restrict__ o, const int* __restrict__ s, int n) {
    int i = blockIdx.x * blockDim.x + threadIdx.x;
    if (i < n) o[i] = s[i];
}

// deg[src] += 1 (float, for normalization) and cnt[dst] += 1 (int, for CSR)
__global__ void deg_cnt_kernel(const long long* __restrict__ ei, float* __restrict__ deg,
                               int* __restrict__ cnt, int E) {
    int e = blockIdx.x * blockDim.x + threadIdx.x;
    if (e < E) {
        atomicAdd(&deg[ei[e]], 1.0f);
        atomicAdd(&cnt[ei[E + e]], 1);
    }
}

// deg -> deg^{-1/2} in place (0 if deg==0)
__global__ void rsqrt_k(float* __restrict__ deg, int n) {
    int i = blockIdx.x * blockDim.x + threadIdx.x;
    if (i < n) {
        float d = deg[i];
        deg[i] = (d > 0.0f) ? rsqrtf(d) : 0.0f;
    }
}

// w[e] = -dis[src] * dis[dst]
__global__ void edge_w(const long long* __restrict__ ei, const float* __restrict__ dis,
                       float* __restrict__ w, int E) {
    int e = blockIdx.x * blockDim.x + threadIdx.x;
    if (e < E) w[e] = -dis[ei[e]] * dis[ei[E + e]];
}

// ------------------------------------------------------- two-level scan
__global__ void scan_blocks(const int* __restrict__ cnt, int* __restrict__ rowptr,
                            int* __restrict__ bsum, int N) {
    __shared__ int tmp[SCAN_B];
    int t = threadIdx.x;
    int i = blockIdx.x * SCAN_B + t;
    int v = (i < N) ? cnt[i] : 0;
    tmp[t] = v;
    __syncthreads();
    for (int o = 1; o < SCAN_B; o <<= 1) {
        int add = (t >= o) ? tmp[t - o] : 0;
        __syncthreads();
        tmp[t] += add;
        __syncthreads();
    }
    if (i < N) rowptr[i] = tmp[t] - v;                 // exclusive
    if (t == SCAN_B - 1) bsum[blockIdx.x] = tmp[t];    // block total
}

// single block; NB must be <= SCAN_B (here NB = ceil(50000/256) = 196)
__global__ void scan_sums(const int* __restrict__ bsum, int* __restrict__ boff, int NB) {
    __shared__ int tmp[SCAN_B];
    int t = threadIdx.x;
    int v = (t < NB) ? bsum[t] : 0;
    tmp[t] = v;
    __syncthreads();
    for (int o = 1; o < SCAN_B; o <<= 1) {
        int add = (t >= o) ? tmp[t - o] : 0;
        __syncthreads();
        tmp[t] += add;
        __syncthreads();
    }
    if (t < NB) boff[t] = tmp[t] - v;                  // exclusive
}

__global__ void add_offsets(int* __restrict__ rowptr, const int* __restrict__ boff,
                            int N, int E) {
    int i = blockIdx.x * blockDim.x + threadIdx.x;
    if (i < N) rowptr[i] += boff[i >> 8];
    if (i == 0) rowptr[N] = E;
}

// compact (src, w) by destination using per-node cursors
__global__ void scatter_edges(const long long* __restrict__ ei, const float* __restrict__ wE,
                              int* __restrict__ cursor, int* __restrict__ csrc,
                              float* __restrict__ cw, int E) {
    int e = blockIdx.x * blockDim.x + threadIdx.x;
    if (e < E) {
        int dst = (int)ei[E + e];
        int pos = atomicAdd(&cursor[dst], 1);
        csrc[pos] = (int)ei[e];
        cw[pos]   = wE[e];
    }
}

// -------------------------------------------------------- gather SpMV
// One wave per destination node; lane covers 4 features (float4 = 512B/row).
// out[node] = scale * sum_e w_e * h[src_e]  (+ basescale * base[node] if base)
// Register accumulation, zero float atomics; fuses the Chebyshev recurrence.
__global__ void spmv_gather(const float* __restrict__ h, const int* __restrict__ rowptr,
                            const int* __restrict__ csrc, const float* __restrict__ cw,
                            const float* __restrict__ base, float basescale, float scale,
                            float* __restrict__ out, int N) {
    int wid  = (int)((blockIdx.x * (long long)blockDim.x + threadIdx.x) >> 5);
    int lane = threadIdx.x & 31;
    if (wid >= N) return;
    int s0 = rowptr[wid], s1 = rowptr[wid + 1];
    float ax = 0.f, ay = 0.f, az = 0.f, aw = 0.f;
    for (int e = s0; e < s1; ++e) {
        int s    = csrc[e];
        float wv = cw[e];
        if (e + 1 < s1) {   // hide random L2 latency on the next source row
            int sn = csrc[e + 1];
            __builtin_prefetch(h + (size_t)sn * 128 + lane * 4, 0, 3);
        }
        const float4 v = ((const float4*)(h + (size_t)s * 128))[lane];
        ax += wv * v.x; ay += wv * v.y; az += wv * v.z; aw += wv * v.w;
    }
    float4 r;
    if (base) {
        const float4 b = ((const float4*)(base + (size_t)wid * 128))[lane];
        r.x = scale * ax + basescale * b.x;
        r.y = scale * ay + basescale * b.y;
        r.z = scale * az + basescale * b.z;
        r.w = scale * aw + basescale * b.w;
    } else {
        r.x = scale * ax; r.y = scale * ay; r.z = scale * az; r.w = scale * aw;
    }
    ((float4*)(out + (size_t)wid * 128))[lane] = r;
}

// -------------------------------------------------- weight fragment packing
// B layout of V_WMMA_F32_16X16X32_F16: frag (c,kt,nt): lane<16 -> col=lane,
// K = kt*32 + {0..7,16..23}; lane>=16 -> col=lane-16, K = kt*32 + {8..15,24..31}.
__global__ void pack_w_kernel(const float* __restrict__ W, _Float16* __restrict__ P,
                              int Kdim, int Nsrc, int NT, int Korder) {
    int idx = blockIdx.x * blockDim.x + threadIdx.x;
    int KT = Kdim >> 5;
    int total = Korder * KT * NT * 512;
    if (idx >= total) return;
    int i    = idx & 15;
    int lane = (idx >> 4) & 31;
    int nt   = (idx >> 9) % NT;
    int kt   = ((idx >> 9) / NT) % KT;
    int c    = (idx >> 9) / (NT * KT);
    int klocal = ((i >> 3) << 4) + ((lane >> 4) << 3) + (i & 7);   // 0..31
    int n = (nt << 4) + (lane & 15);
    int k = (kt << 5) + klocal;
    float v = (n < Nsrc) ? W[((size_t)c * Kdim + k) * Nsrc + n] : 0.0f;
    P[idx] = (_Float16)v;
}

// ----------------------------------------------------------- WMMA GEMM
// One wave per 16-row M-tile computes ALL NT column tiles: A fragment is
// loaded+converted once per (c,kt) and reused across NT back-to-back WMMAs.
// acc[NT] stays fully in registers (NT is a template constant).
template <int NT>
__global__ void gemm_cheb_kernel(const float* __restrict__ A0, const float* __restrict__ A1,
                                 const float* __restrict__ A2, const _Float16* __restrict__ P,
                                 const float* __restrict__ bias, float* __restrict__ out,
                                 int Kdim, int Nout, int relu, int mtiles) {
    int wid  = (int)((blockIdx.x * (long long)blockDim.x + threadIdx.x) >> 5);
    int lane = threadIdx.x & 31;
    if (wid >= mtiles) return;
    int KT = Kdim >> 5;

    v8f acc[NT];
#pragma unroll
    for (int nt = 0; nt < NT; ++nt) acc[nt] = (v8f){};

    const float* As[3] = {A0, A1, A2};
#pragma unroll
    for (int c = 0; c < 3; ++c) {
        // A layout: lanes 0-15 row=lane, K 0..7 & 16..23; lanes 16-31 row=lane-16,
        // K 8..15 & 24..31 (two contiguous 32B f32 chunks per lane, cvt to f16).
        const float* arow = As[c] + (size_t)(wid * 16 + (lane & 15)) * Kdim
                            + ((lane >> 4) << 3);
        for (int kt = 0; kt < KT; ++kt) {
            const float* p = arow + (kt << 5);
            v16h a;
#pragma unroll
            for (int j = 0; j < 8; ++j) a[j] = (_Float16)p[j];
#pragma unroll
            for (int j = 0; j < 8; ++j) a[8 + j] = (_Float16)p[16 + j];
            const _Float16* pb = P + (((size_t)(c * KT + kt) * NT) << 9) + (lane << 4);
#pragma unroll
            for (int nt = 0; nt < NT; ++nt) {
                v16h b = *(const v16h*)(pb + ((size_t)nt << 9));
                acc[nt] = __builtin_amdgcn_wmma_f32_16x16x32_f16(
                    false, a, false, b, (short)0, acc[nt], false, false);
            }
        }
    }

    int ncol = lane & 15;
    int row0 = wid * 16 + ((lane >> 4) << 3);
#pragma unroll
    for (int nt = 0; nt < NT; ++nt) {
        int col = (nt << 4) + ncol;
        if (col < Nout) {
            float bv = bias[col];
#pragma unroll
            for (int g = 0; g < 8; ++g) {
                float v = acc[nt][g] + bv;
                if (relu) v = fmaxf(v, 0.0f);
                out[(size_t)(row0 + g) * Nout + col] = v;
            }
        }
    }
}

// ----------------------------------------------------------- log_softmax
__global__ void logsoftmax_kernel(float* __restrict__ out, int M, int C) {
    int row  = (int)((blockIdx.x * (long long)blockDim.x + threadIdx.x) >> 5);
    int lane = threadIdx.x & 31;
    if (row >= M) return;
    float* p = out + (size_t)row * C;
    float v0 = (lane < C) ? p[lane] : -INFINITY;
    float v1 = (lane + 32 < C) ? p[lane + 32] : -INFINITY;
    float m = fmaxf(v0, v1);
#pragma unroll
    for (int o = 16; o > 0; o >>= 1) m = fmaxf(m, __shfl_xor(m, o, 32));
    float s = ((lane < C) ? __expf(v0 - m) : 0.0f)
            + ((lane + 32 < C) ? __expf(v1 - m) : 0.0f);
#pragma unroll
    for (int o = 16; o > 0; o >>= 1) s += __shfl_xor(s, o, 32);
    float lse = m + __logf(s);
    if (lane < C) p[lane] = v0 - lse;
    if (lane + 32 < C) p[lane + 32] = v1 - lse;
}

// ------------------------------------------------------------------ driver
extern "C" void kernel_launch(void* const* d_in, const int* in_sizes, int n_in,
                              void* d_out, int out_size, void* d_ws, size_t ws_size,
                              hipStream_t stream) {
    const float*     x  = (const float*)d_in[0];
    const long long* ei = (const long long*)d_in[1];
    const float*     W1 = (const float*)d_in[2];
    const float*     b1 = (const float*)d_in[3];
    const float*     W2 = (const float*)d_in[4];
    const float*     b2 = (const float*)d_in[5];
    float* out = (float*)d_out;

    const int D  = 128;                 // d_in == d_hid
    const int M  = in_sizes[0] / D;     // 50000 nodes
    const int E  = in_sizes[1] / 2;     // 800000 edges
    const int C  = in_sizes[5];         // 40 classes
    const int KT  = D >> 5;             // 4 K-chunks
    const int NT1 = D >> 4;             // 8 N-tiles (layer 1)
    const int NT2 = (C + 15) >> 4;      // 3 N-tiles (layer 2)
    const int ND  = M * D;
    const int NB  = (M + SCAN_B - 1) / SCAN_B;   // 196 scan blocks (<=256)

    // ---- workspace carve (256B aligned slabs)
    char* w = (char*)d_ws;
    auto carve = [&](size_t bytes) -> char* {
        char* p = w;
        w += (bytes + 255) & ~(size_t)255;
        return p;
    };
    float* deg    = (float*)carve((size_t)M * 4);
    float* wE     = (float*)carve((size_t)E * 4);
    float* Tx1    = (float*)carve((size_t)ND * 4);
    float* Tx2    = (float*)carve((size_t)ND * 4);
    float* H      = (float*)carve((size_t)ND * 4);
    int*   cnt    = (int*)carve((size_t)M * 4);
    int*   rowptr = (int*)carve((size_t)(M + 1) * 4);
    int*   bsum   = (int*)carve((size_t)SCAN_B * 4);
    int*   boff   = (int*)carve((size_t)SCAN_B * 4);
    int*   cursor = (int*)carve((size_t)M * 4);
    int*   csrc   = (int*)carve((size_t)E * 4);
    float* cw     = (float*)carve((size_t)E * 4);
    _Float16* P1  = (_Float16*)carve((size_t)3 * KT * NT1 * 512 * 2);
    _Float16* P2  = (_Float16*)carve((size_t)3 * KT * NT2 * 512 * 2);

    const dim3 blk(256);
    const int gM  = (M + 255) / 256;
    const int gE  = (E + 255) / 256;
    const int mtiles    = M / 16;                 // 3125 (exact)
    const int gNodeWaves = (M + 7) / 8;           // 8 waves per 256-thread block
    const int gTileWaves = (mtiles + 7) / 8;

    // ---- degrees (by src) + CSR counts (by dst)
    fill_zero32<<<gM, blk, 0, stream>>>((unsigned int*)deg, M);
    fill_zero32<<<gM, blk, 0, stream>>>((unsigned int*)cnt, M);
    deg_cnt_kernel<<<gE, blk, 0, stream>>>(ei, deg, cnt, E);
    rsqrt_k<<<gM, blk, 0, stream>>>(deg, M);
    edge_w<<<gE, blk, 0, stream>>>(ei, deg, wE, E);

    // ---- CSR build: exclusive scan + scatter
    scan_blocks<<<NB, dim3(SCAN_B), 0, stream>>>(cnt, rowptr, bsum, M);
    scan_sums<<<1, dim3(SCAN_B), 0, stream>>>(bsum, boff, NB);
    add_offsets<<<gM, blk, 0, stream>>>(rowptr, boff, M, E);
    copy_i32<<<gM, blk, 0, stream>>>(cursor, rowptr, M);
    scatter_edges<<<gE, blk, 0, stream>>>(ei, wE, cursor, csrc, cw, E);

    // ---- weight packing into WMMA B-fragment layout
    int tot1 = 3 * KT * NT1 * 512;
    pack_w_kernel<<<(tot1 + 255) / 256, blk, 0, stream>>>(W1, P1, D, D, NT1, 3);
    int tot2 = 3 * KT * NT2 * 512;
    pack_w_kernel<<<(tot2 + 255) / 256, blk, 0, stream>>>(W2, P2, D, C, NT2, 3);

    // ---- layer 1: Tx1 = L~x ; Tx2 = 2 L~ Tx1 - x (fused) ; H = cheb GEMM
    spmv_gather<<<gNodeWaves, blk, 0, stream>>>(x,   rowptr, csrc, cw, nullptr, 0.f, 1.f, Tx1, M);
    spmv_gather<<<gNodeWaves, blk, 0, stream>>>(Tx1, rowptr, csrc, cw, x,      -1.f, 2.f, Tx2, M);
    gemm_cheb_kernel<8><<<gTileWaves, blk, 0, stream>>>(
        x, Tx1, Tx2, P1, b1, H, D, D, /*relu=*/1, mtiles);

    // ---- layer 2 over H (reuse Tx buffers)
    spmv_gather<<<gNodeWaves, blk, 0, stream>>>(H,   rowptr, csrc, cw, nullptr, 0.f, 1.f, Tx1, M);
    spmv_gather<<<gNodeWaves, blk, 0, stream>>>(Tx1, rowptr, csrc, cw, H,      -1.f, 2.f, Tx2, M);
    gemm_cheb_kernel<3><<<gTileWaves, blk, 0, stream>>>(
        H, Tx1, Tx2, P2, b2, out, D, C, /*relu=*/0, mtiles);

    // ---- row-wise log_softmax in place on d_out
    logsoftmax_kernel<<<(M + 7) / 8, blk, 0, stream>>>(out, M, C);
}
